// MyVQmodel_87342454931977
// MI455X (gfx1250) — matrix operand order repaired and checked by reading, not compiled, not run on hardware
//
#include <hip/hip_runtime.h>
#include <hip/hip_bf16.h>
#include <math.h>

// ---------------------------------------------------------------------------
// VQ-VAE vector quantization for MI455X (gfx1250, wave32, WMMA).
//   z:      [4, 256, 16,16,16] f32   -> zf [16384, 256] (row r = b*4096 + s,
//            element zf[r,c] = z[b*1048576 + c*4096 + s])
//   weight: [8192, 256] f32
// Distance argmin == argmax over n of  s(r,n) = 2*z_r.w_n - |w_n|^2.
// Matmul (68.7 GFLOP) runs on v_wmma_f32_16x16x32_bf16 with f32 accum.
// Weight chunks are staged LDS-side with global_load_async_to_lds_b128
// (ASYNCcnt DMA path), double-buffered, with the wait overlapped by compute.
// ---------------------------------------------------------------------------

typedef __attribute__((ext_vector_type(16))) __bf16 v16bf;
typedef __attribute__((ext_vector_type(8)))  __bf16 v8bf;
typedef __attribute__((ext_vector_type(8)))  float  v8f;

union ABu { v16bf v; v8bf h[2]; };

#define NE_CODES 8192
#define CDIM     256
#define NROWS    16384
#define MTILE    64          // z-rows per workgroup in the argmin kernel
#define LPAD     264         // LDS row stride in bf16 (256 + 8 pad, 528B = 33*16B)

#define WMMA_BF16(A, B, C) __builtin_amdgcn_wmma_f32_16x16x32_bf16( \
    false, (A), false, (B), (short)0, (C), false, false)

// ---------------------------------------------------------------------------
// Kernel 0: weight -> bf16 copy, |w|^2 per code, zero histogram.
// grid = 8192 blocks x 256 threads (one block per code row).
// ---------------------------------------------------------------------------
__global__ __launch_bounds__(256) void vq_prep_kernel(
    const float* __restrict__ weight,
    unsigned short* __restrict__ wbf_raw,   // bf16 storage
    float* __restrict__ wsq,
    float* __restrict__ counts)
{
    const int j = blockIdx.x;
    const int t = threadIdx.x;
    __bf16* wbf = (__bf16*)wbf_raw;

    float v = weight[j * CDIM + t];
    wbf[j * CDIM + t] = (__bf16)v;          // RNE f32->bf16

    __shared__ float red[256];
    red[t] = v * v;
    __syncthreads();
    #pragma unroll
    for (int s = 128; s > 0; s >>= 1) {
        if (t < s) red[t] += red[t + s];
        __syncthreads();
    }
    if (t == 0) wsq[j] = red[0];

    if (j < NE_CODES / 256) counts[j * 256 + t] = 0.0f;   // zero 8192 counts
}

// ---------------------------------------------------------------------------
// Kernel 1: WMMA distance + argmin.
// grid = 256 blocks (64 rows each) x 256 threads (8 waves).
//   wave = 4 M-subtiles x 2 N-subtiles; loops 256 chunks of 32 codes.
// ---------------------------------------------------------------------------

// Async DMA of one 32-code x 256-ch bf16 chunk (16 KB) into LDS.
// 256 threads x 64B each; INST_OFFSET applies to both LDS and global address,
// so one LDS-base VGPR + offset:0/16/32/48 covers the thread's 64B run.
__device__ __forceinline__ void load_w_chunk_async(
    __bf16 (*dst)[LPAD], const __bf16* __restrict__ wbf, int nc, int tid)
{
    const int row  = tid >> 3;          // 0..31
    const int coff = (tid & 7) * 32;    // 0..224
    unsigned lds = (unsigned)(uintptr_t)(&dst[row][coff]);
    unsigned long long g =
        (unsigned long long)(uintptr_t)(wbf + (nc + row) * CDIM + coff);
    asm volatile(
        "global_load_async_to_lds_b128 %0, %1, off\n\t"
        "global_load_async_to_lds_b128 %0, %1, off offset:16\n\t"
        "global_load_async_to_lds_b128 %0, %1, off offset:32\n\t"
        "global_load_async_to_lds_b128 %0, %1, off offset:48"
        :: "v"(lds), "v"(g) : "memory");
    // Prefetch the chunk after next into GL2 (global_prefetch_b8).
    if (nc + 32 < NE_CODES)
        __builtin_prefetch(wbf + (nc + 32 + row) * CDIM + coff, 0, 1);
}

#define WAIT_ASYNC0() asm volatile("s_wait_asynccnt 0x0" ::: "memory")

__global__ __launch_bounds__(256) void vq_argmin_kernel(
    const float* __restrict__ z,
    const unsigned short* __restrict__ wbf_raw,
    const float* __restrict__ wsq,
    int*   __restrict__ idx_int,
    float* __restrict__ idx_out)
{
    __shared__ __align__(16) __bf16 zb[MTILE][LPAD];
    __shared__ __align__(16) __bf16 wb[2][32][LPAD];
    __shared__ float rv[MTILE][32];
    __shared__ int   ri[MTILE][32];

    const __bf16* wbf = (const __bf16*)wbf_raw;
    const int tid    = threadIdx.x;
    const int blk    = blockIdx.x;
    const int r0     = blk * MTILE;          // first z-row of this block
    const int b      = r0 >> 12;             // batch (4096 spatials per batch)
    const int s_base = r0 & 4095;

    // ---- stage first weight chunk via async DMA (overlaps z staging) -----
    load_w_chunk_async(wb[0], wbf, 0, tid);

    // ---- stage z tile (64 rows x 256 ch) into LDS as bf16 ----------------
    {
        const int s_local = tid & 63;
        const int cg      = tid >> 6;        // 4 channel groups
        const float* zp = z + (size_t)b * 1048576 + s_base + s_local;
        #pragma unroll 4
        for (int i = 0; i < 64; ++i) {
            int c = i * 4 + cg;
            zb[s_local][c] = (__bf16)zp[(size_t)c * 4096];  // coalesced reads
        }
    }
    WAIT_ASYNC0();
    __syncthreads();

    const int lane  = tid & 31;
    const int wave  = tid >> 5;
    const int mtile = wave & 3;              // which 16-row M subtile
    const int ntile = wave >> 2;             // which 16-code N subtile of chunk
    const int half  = lane >> 4;
    const int l16   = lane & 15;

    // ---- preload all 8 A fragments (this wave's 16 z-rows, K=256) --------
    // A K order chosen here (contiguous; B uses the same): lane holds row M=lane%16; half selects K half:
    //   lo 8 elems: K = kk*32 + 16*half + (0..7)
    //   hi 8 elems: K = kk*32 + 16*half + 8 + (0..7)
    ABu afrag[8];
    {
        const __bf16* zrow = &zb[mtile * 16 + l16][0];
        #pragma unroll
        for (int kk = 0; kk < 8; ++kk) {
            afrag[kk].h[0] = *(const v8bf*)(zrow + kk * 32 + 16 * half);
            afrag[kk].h[1] = *(const v8bf*)(zrow + kk * 32 + 16 * half + 8);
        }
    }

    float best[8];
    int   bidx[8];
    #pragma unroll
    for (int v = 0; v < 8; ++v) { best[v] = -3.402823e38f; bidx[v] = 0; }

    // ---- main loop over 8192 codes in chunks of 32 -----------------------
    for (int nc = 0; nc < NE_CODES; nc += 32) {
        const int cur = (nc >> 5) & 1;
        if (nc + 32 < NE_CODES)
            load_w_chunk_async(wb[1 - cur], wbf, nc + 32, tid);

        const int n  = nc + ntile * 16 + l16;   // this lane's code column
        const float wq = wsq[n];

        v8f acc;
        #pragma unroll
        for (int v = 0; v < 8; ++v) acc[v] = 0.0f;

        // B layout (32x16 bf16): lane holds column N=lane%16,
        // fragment kk occupies K = kk*32 + 16*half + (0..15): 32B contiguous.
        // Software-pipelined with two B buffers so DS latency hides under the
        // matrix pipe instead of a full s_wait_dscnt 0 before every WMMA.
        const __bf16* wbase = &wb[cur][ntile * 16 + l16][16 * half];
        ABu bA, bB;
        #define LOADB(dst, kk) \
            dst.h[0] = *(const v8bf*)(wbase + (kk) * 32); \
            dst.h[1] = *(const v8bf*)(wbase + (kk) * 32 + 8)
        LOADB(bA, 0);
        LOADB(bB, 1);
        acc = WMMA_BF16(afrag[0].v, bA.v, acc);
        LOADB(bA, 2);
        acc = WMMA_BF16(afrag[1].v, bB.v, acc);
        LOADB(bB, 3);
        acc = WMMA_BF16(afrag[2].v, bA.v, acc);
        LOADB(bA, 4);
        acc = WMMA_BF16(afrag[3].v, bB.v, acc);
        LOADB(bB, 5);
        acc = WMMA_BF16(afrag[4].v, bA.v, acc);
        LOADB(bA, 6);
        acc = WMMA_BF16(afrag[5].v, bB.v, acc);
        LOADB(bB, 7);
        acc = WMMA_BF16(afrag[6].v, bA.v, acc);
        acc = WMMA_BF16(afrag[7].v, bB.v, acc);
        #undef LOADB

        // score = 2*z.w - |w|^2 ; running argmax (strict > keeps lowest n)
        #pragma unroll
        for (int v = 0; v < 8; ++v) {
            float s = 2.0f * acc[v] - wq;
            if (s > best[v]) { best[v] = s; bidx[v] = n; }
        }

        // Publish next chunk's buffer: each wave drains its own async DMA,
        // then the barrier makes the whole 16KB chunk visible to all waves.
        WAIT_ASYNC0();
        __syncthreads();
    }

    // ---- cross-lane / cross-wave argmax reduce ---------------------------
    // C/D 16x16 layout: lanes 0-15 hold M=v, lanes 16-31 hold M=v+8.
    #pragma unroll
    for (int v = 0; v < 8; ++v) {
        int row  = mtile * 16 + v + 8 * half;
        int slot = ntile * 16 + l16;
        rv[row][slot] = best[v];
        ri[row][slot] = bidx[v];
    }
    __syncthreads();
    if (tid < MTILE) {
        float bv = rv[tid][0];
        int   bi = ri[tid][0];
        #pragma unroll 8
        for (int j = 1; j < 32; ++j) {
            float v = rv[tid][j];
            int   i2 = ri[tid][j];
            if (v > bv || (v == bv && i2 < bi)) { bv = v; bi = i2; }
        }
        idx_int[r0 + tid] = bi;
        idx_out[r0 + tid] = (float)bi;   // reference idx output, as f32
    }
}

// ---------------------------------------------------------------------------
// Kernel 2: gather z_q, emit out = z + (q - z)  (straight-through, replicated
// literally for bit-faithfulness), per-block loss partials, histogram.
// grid = 256 blocks (64 rows) x 256 threads.
// ---------------------------------------------------------------------------
__global__ __launch_bounds__(256) void vq_gather_kernel(
    const float* __restrict__ z,
    const float* __restrict__ weight,
    const int*   __restrict__ idx_int,
    float* __restrict__ out,
    float* __restrict__ counts,
    float* __restrict__ loss_part)
{
    const int tid = threadIdx.x;
    const int blk = blockIdx.x;
    const int r0  = blk * MTILE;
    const int b   = r0 >> 12;
    const int s_base  = r0 & 4095;
    const int s_local = tid & 63;
    const int cg      = tid >> 6;

    const int id = idx_int[r0 + s_local];
    const float* wrow = weight + (size_t)id * CDIM;
    const float* zp = z   + (size_t)b * 1048576 + s_base + s_local;
    float*       op = out + (size_t)b * 1048576 + s_base + s_local;

    float lsum = 0.0f;
    #pragma unroll 4
    for (int i = 0; i < 64; ++i) {
        int c = i * 4 + cg;
        float q  = wrow[c];                  // gather: 8MB table lives in L2
        float zv = zp[(size_t)c * 4096];     // coalesced along spatial axis
        float d  = q - zv;
        lsum += d * d;
        op[(size_t)c * 4096] = zv + d;       // == z + stopgrad(q - z)
    }

    __shared__ float red[256];
    red[tid] = lsum;
    __syncthreads();
    #pragma unroll
    for (int s = 128; s > 0; s >>= 1) {
        if (tid < s) red[tid] += red[tid + s];
        __syncthreads();
    }
    if (tid == 0) loss_part[blk] = red[0];   // deterministic per-block partial

    // one +1.0f per row; integer-valued float adds are order-independent
    if (cg == 0) atomicAdd(&counts[id], 1.0f);
}

// ---------------------------------------------------------------------------
// Kernel 3: loss = BETA * mean(d^2), perplexity = exp(-sum p log(p+1e-10)).
// ---------------------------------------------------------------------------
__global__ __launch_bounds__(256) void vq_finalize_kernel(
    const float* __restrict__ counts,
    const float* __restrict__ loss_part,
    float* __restrict__ out_scalars)      // [0]=loss, [1]=perplexity
{
    const int tid = threadIdx.x;
    float h = 0.0f;
    #pragma unroll 8
    for (int j = 0; j < NE_CODES / 256; ++j) {
        float p = counts[tid + j * 256] * (1.0f / (float)NROWS);
        h += p * logf(p + 1e-10f);
    }
    float l = loss_part[tid];

    __shared__ float rh[256], rl[256];
    rh[tid] = h; rl[tid] = l;
    __syncthreads();
    #pragma unroll
    for (int s = 128; s > 0; s >>= 1) {
        if (tid < s) { rh[tid] += rh[tid + s]; rl[tid] += rl[tid + s]; }
        __syncthreads();
    }
    if (tid == 0) {
        out_scalars[0] = 0.25f * rl[0] / 4194304.0f;   // BETA * mean
        out_scalars[1] = expf(-rh[0]);
    }
}

// ---------------------------------------------------------------------------
// Launch. Output layout (flat f32): out[4194304] | loss | perplexity | idx[16384]
// Workspace: bf16 weight (4MB) | wsq (32KB) | counts (32KB) | loss_part (1KB)
//            | idx_int (64KB)
// ---------------------------------------------------------------------------
extern "C" void kernel_launch(void* const* d_in, const int* in_sizes, int n_in,
                              void* d_out, int out_size, void* d_ws, size_t ws_size,
                              hipStream_t stream) {
    const float* z      = (const float*)d_in[0];
    const float* weight = (const float*)d_in[1];
    float* out = (float*)d_out;

    char* ws = (char*)d_ws;
    unsigned short* wbf = (unsigned short*)ws;                      // 4 MB bf16
    float* wsq       = (float*)(ws + (4u << 20));                   // 32 KB
    float* counts    = (float*)(ws + (4u << 20) + 32768);           // 32 KB
    float* loss_part = (float*)(ws + (4u << 20) + 65536);           // 1 KB
    int*   idx_int   = (int*)  (ws + (4u << 20) + 65536 + 1024);    // 64 KB

    float* out_scalars = out + 4194304;       // loss, perplexity
    float* idx_out     = out + 4194306;       // 16384 indices as f32

    vq_prep_kernel<<<NE_CODES, 256, 0, stream>>>(weight, wbf, wsq, counts);
    vq_argmin_kernel<<<NROWS / MTILE, 256, 0, stream>>>(z, wbf, wsq,
                                                        idx_int, idx_out);
    vq_gather_kernel<<<NROWS / MTILE, 256, 0, stream>>>(z, weight, idx_int,
                                                        out, counts, loss_part);
    vq_finalize_kernel<<<1, 256, 0, stream>>>(counts, loss_part, out_scalars);
}
